// FeatureEmbeddings_single_62912680952641
// MI455X (gfx1250) — compile-verified, matrix operand order checked by reading, not run on hardware
//
#include <hip/hip_runtime.h>
#include <hip/hip_bf16.h>

typedef _Float16 v8h  __attribute__((ext_vector_type(8)));
typedef _Float16 v16h __attribute__((ext_vector_type(16)));
typedef float    v8f  __attribute__((ext_vector_type(8)));

// Fused  out = relu( relu(x ⊗ W1 + b1) @ W2^T + b2 )
// M = B*F = 179200 rows, K = H = 512, N = D = 256.
// Block: 256 threads = 8 waves. Block tile: 256(M) x 64(N).
// Wave w: rows [mBase + 32w, mBase + 32w + 32), all 4 N-tiles of the block.
// Per k-step: 2 A-frags (generated on the fly), 4 B-frags (LDS), 8 WMMAs.
__global__ __launch_bounds__(256)
void mlp2_wmma_f16(const float* __restrict__ x,
                   const float* __restrict__ W1,
                   const float* __restrict__ b1,
                   const float* __restrict__ W2,
                   const float* __restrict__ b2,
                   float* __restrict__ out)
{
    constexpr int H  = 512;   // K
    constexpr int D  = 256;   // N total
    constexpr int NT = 4;     // N-tiles per block (block N = 64)

    // LDS: W2 tile in f16, pre-swizzled to per-lane B-fragment order.
    // sB[kt][nt][lane][j], kt in [0,16), nt in [0,4), lane in [0,32), j in [0,16)
    __shared__ _Float16 sW1[H];
    __shared__ _Float16 sB1[H];
    __shared__ _Float16 sB[16 * NT * 32 * 16];   // 64 KB

    const int tid   = threadIdx.x;
    const int wave  = tid >> 5;
    const int lane  = tid & 31;
    const int lr    = lane & 15;    // row/col within 16
    const int khalf = lane >> 4;    // K-half selector in A/B fragment layout

    const int mBase = blockIdx.x * 256;
    const int nBase = blockIdx.y * (NT * 16);

    // ---- stage W1 / b1 as f16 ----
    for (int i = tid; i < H; i += 256) {
        sW1[i] = (_Float16)W1[i];   // W1 is [H,1] -> flat H
        sB1[i] = (_Float16)b1[i];
    }
    // ---- stage W2 tile (64 cols x 512 k) as f16, swizzled to fragment order ----
    // B-fragment element j of lane L (kh = L>>4) maps to
    //   k_local = j + 8*kh + (j>=8 ? 8 : 0)   within the 32-wide k-tile,
    //   n       = nBase + nt*16 + (L&15).
    for (int i = tid; i < 16 * NT * 32 * 16; i += 256) {
        const int j  = i & 15;
        const int L  = (i >> 4) & 31;
        const int nt = (i >> 9) & (NT - 1);
        const int kt = i >> 11;
        const int kh = L >> 4;
        const int klocal = j + 8 * kh + ((j >= 8) ? 8 : 0);
        const int k = kt * 32 + klocal;
        const int n = nBase + nt * 16 + (L & 15);
        sB[i] = (_Float16)W2[n * H + k];        // W2 is [D,H] row-major
    }
    __syncthreads();

    // ---- per-lane scalars ----
    const int rowBase = mBase + wave * 32;
    const _Float16 xh0 = (_Float16)x[rowBase + lr];        // rows of m-tile 0
    const _Float16 xh1 = (_Float16)x[rowBase + 16 + lr];   // rows of m-tile 1
    float b2v[NT];
    #pragma unroll
    for (int nt = 0; nt < NT; ++nt) b2v[nt] = b2[nBase + nt * 16 + lr];

    v16h x0v, x1v;
    #pragma unroll
    for (int e = 0; e < 16; ++e) { x0v[e] = xh0; x1v[e] = xh1; }

    v8f acc0[NT], acc1[NT];
    #pragma unroll
    for (int nt = 0; nt < NT; ++nt) { acc0[nt] = {}; acc1[nt] = {}; }

    // ---- K loop: 16 steps of K=32, 8 WMMAs per step ----
    #pragma unroll
    for (int kt = 0; kt < 16; ++kt) {
        const int kb = kt * 32 + khalf * 8;

        union HV { v16h v; v8h h[2]; };
        HV w1v, b1v;
        w1v.h[0] = *(const v8h*)&sW1[kb];
        w1v.h[1] = *(const v8h*)&sW1[kb + 16];
        b1v.h[0] = *(const v8h*)&sB1[kb];
        b1v.h[1] = *(const v8h*)&sB1[kb + 16];

        v16h bf[NT];
        #pragma unroll
        for (int nt = 0; nt < NT; ++nt)
            bf[nt] = *(const v16h*)&sB[((kt * NT + nt) * 32 + lane) * 16];

        const v16h zero = {};
        // A fragments generated on the fly: h = relu(x*W1 + b1), packed f16 math
        v16h a0 = __builtin_elementwise_max(x0v * w1v.v + b1v.v, zero);
        v16h a1 = __builtin_elementwise_max(x1v * w1v.v + b1v.v, zero);

        #pragma unroll
        for (int nt = 0; nt < NT; ++nt) {
            acc0[nt] = __builtin_amdgcn_wmma_f32_16x16x32_f16(
                false, a0, false, bf[nt], (short)0, acc0[nt], false, false);
            acc1[nt] = __builtin_amdgcn_wmma_f32_16x16x32_f16(
                false, a1, false, bf[nt], (short)0, acc1[nt], false, false);
        }
    }

    // ---- epilogue: +b2, relu, store f32 ----
    // C/D layout: lane (0-15 | 16-31) -> col = lane&15, VGPR r -> row r + 8*khalf
    const int row0 = rowBase + khalf * 8;
    #pragma unroll
    for (int nt = 0; nt < NT; ++nt) {
        const int col = nBase + nt * 16 + lr;
        #pragma unroll
        for (int r = 0; r < 8; ++r) {
            out[(row0 + r) * D + col]      = fmaxf(acc0[nt][r] + b2v[nt], 0.0f);
            out[(row0 + 16 + r) * D + col] = fmaxf(acc1[nt][r] + b2v[nt], 0.0f);
        }
    }
}

extern "C" void kernel_launch(void* const* d_in, const int* in_sizes, int n_in,
                              void* d_out, int out_size, void* d_ws, size_t ws_size,
                              hipStream_t stream) {
    const float* x  = (const float*)d_in[0];   // [B,F]  = 179200 floats
    const float* W1 = (const float*)d_in[1];   // [H,1]  = 512
    const float* b1 = (const float*)d_in[2];   // [H]    = 512
    const float* W2 = (const float*)d_in[3];   // [D,H]  = 256*512
    const float* b2 = (const float*)d_in[4];   // [D]    = 256
    float* out = (float*)d_out;                // [B,F,D] = M*256

    const int M = in_sizes[0];                 // 179200, divisible by 256
    dim3 grid(M / 256, 256 / 64, 1);           // (700, 4)
    dim3 block(256, 1, 1);
    mlp2_wmma_f16<<<grid, block, 0, stream>>>(x, W1, b1, W2, b2, out);
}
